// AlternatingForecastModel_21414706938084
// MI455X (gfx1250) — compile-verified, matrix operand order checked by reading, not run on hardware
//
#include <hip/hip_runtime.h>
#include <stdint.h>

typedef __attribute__((ext_vector_type(16))) __bf16 v16bf;
typedef __attribute__((ext_vector_type(8)))  float  v8f;

#define S_LEN 2688
#define F_IN  32
#define HID   512
#define WEEK  672
#define NG    2048      // 4*HID gate outputs
#define NT    128       // NG/16 n-tiles
#define NK0   17        // layer0 K = 32(cur) + 512(h0) -> 17 k-steps of 32
#define NK1   32        // layer1 K = 512(h0) + 512(h1) -> 32 k-steps
#define STA   1064      // A stride: 32(cur)+512(h0)+512(h1)+8 pad, bf16 elems
#define COL_H0 32
#define COL_H1 544
#define BLOCK 512       // 16 waves; wave w owns hidden tiles {2w, 2w+1} x 4 gates

__device__ __forceinline__ uint16_t f2bf(float f) {
  union { float f; uint32_t u; } v; v.f = f;
  uint32_t u = v.u;
  uint32_t r = u + 0x7fffu + ((u >> 16) & 1u);   // RNE
  return (uint16_t)(r >> 16);
}
__device__ __forceinline__ float bf2f(uint16_t h) {
  union { uint32_t u; float f; } v; v.u = ((uint32_t)h) << 16;
  return v.f;
}
// v_rcp_f32-based fast activations (TRANS ops co-execute with XDL WMMA)
__device__ __forceinline__ float sigm(float x) {
  return __builtin_amdgcn_rcpf(1.f + __expf(-x));
}
__device__ __forceinline__ float tanh_fast(float x) {
  x = fminf(fmaxf(x, -15.f), 15.f);
  float e = __expf(-2.f * x);
  return (1.f - e) * __builtin_amdgcn_rcpf(1.f + e);
}

// ---------------------------------------------------------------------------
// One-shot weight pack: fp32 -> bf16, fragment-major layout.
// Block (ntile, kstep) = 512 bf16 = 1KB; lane L holds W[n, k0 + (L>>4)*16 .. +15]
// with n = ntile*16 + (L&15)  == B-operand per-lane data in contiguous k order (k = 16h + s), as the A loads use.
// Layer0 B = [Wih0 cols 0..31 | Whh0]; Wih0 col 32 (the flag column) is kept
// separately in fp32 (w32s) and folded into the bias at runtime.
// ---------------------------------------------------------------------------
__global__ void pack_kernel(const float* __restrict__ Wih0, const float* __restrict__ Whh0,
                            const float* __restrict__ bih0, const float* __restrict__ bhh0,
                            const float* __restrict__ Wih1, const float* __restrict__ Whh1,
                            const float* __restrict__ bih1, const float* __restrict__ bhh1,
                            uint16_t* __restrict__ w0p, uint16_t* __restrict__ w1p,
                            float* __restrict__ b0s, float* __restrict__ b1s,
                            float* __restrict__ w32s)
{
  const int N0 = NT * NK0 * 512;
  const int N1 = NT * NK1 * 512;
  int idx = blockIdx.x * blockDim.x + threadIdx.x;
  if (idx < N0) {
    int e = idx & 15, lane = (idx >> 4) & 31, blk = idx >> 9;
    int ks = blk % NK0, ntile = blk / NK0;
    int n = ntile * 16 + (lane & 15);
    int k = ks * 32 + ((lane >> 4) << 4) + e;
    float v = (k < 32) ? Wih0[n * (F_IN + 1) + k] : Whh0[n * HID + (k - 32)];
    w0p[idx] = f2bf(v);
  } else if (idx < N0 + N1) {
    int j = idx - N0;
    int e = j & 15, lane = (j >> 4) & 31, blk = j >> 9;
    int ks = blk % NK1, ntile = blk / NK1;
    int n = ntile * 16 + (lane & 15);
    int k = ks * 32 + ((lane >> 4) << 4) + e;
    float v = (k < HID) ? Wih1[n * HID + k] : Whh1[n * HID + (k - HID)];
    w1p[j] = f2bf(v);
  } else if (idx < N0 + N1 + NG) {
    int n = idx - (N0 + N1);
    b0s[n] = bih0[n] + bhh0[n];
  } else if (idx < N0 + N1 + 2 * NG) {
    int n = idx - (N0 + N1 + NG);
    b1s[n] = bih1[n] + bhh1[n];
  } else if (idx < N0 + N1 + 3 * NG) {
    int n = idx - (N0 + N1 + 2 * NG);
    w32s[n] = Wih0[n * (F_IN + 1) + 32];   // flag column
  }
}

// ---------------------------------------------------------------------------
// Persistent per-batch-tile LSTM. Each block owns 16 batch rows and iterates
// all S_LEN steps. 16 waves; wave w owns hidden tiles {2w, 2w+1} x 4 gates
// = 8 WMMA accumulator tiles. Single activation buffer per block:
//   A = [cur(32) | h0(512) | h1(512)]  (bf16, padded stride)
// Layer0 A-matrix = cols 0..543, layer1 A-matrix = cols 32..1055; both GEMMs
// are uniform #pragma unroll 1 k-loops (no loop-invariant weight addresses,
// so nothing gets hoisted/spilled).
// ---------------------------------------------------------------------------
__global__ __launch_bounds__(BLOCK, 1)
void lstm_kernel(const float* __restrict__ x,
                 const uint16_t* __restrict__ w0p,
                 const uint16_t* __restrict__ w1p,
                 const float* __restrict__ b0s,
                 const float* __restrict__ b1s,
                 const float* __restrict__ w32s,
                 const float* __restrict__ Wout,
                 const float* __restrict__ bout,
                 float* __restrict__ out)
{
  __shared__ uint16_t A[16 * STA];
  __shared__ float prev_pred[16];

  const int tid   = threadIdx.x;
  const int wave  = tid >> 5;         // 0..15
  const int lane  = tid & 31;
  const int jl    = lane & 15;
  const int mhalf = lane >> 4;
  const int bbase = blockIdx.x << 4;

  for (int i = tid; i < 16 * STA; i += BLOCK) A[i] = 0;
  if (tid < 16) prev_pred[tid] = 0.f;

  float c0s[2][8], c1s[2][8];
  #pragma unroll
  for (int j = 0; j < 2; ++j)
    #pragma unroll
    for (int r = 0; r < 8; ++r) { c0s[j][r] = 0.f; c1s[j][r] = 0.f; }

  // Bias per accumulator tile: C layout col N = jl, so bias is lane-scalar.
  float bias0[8], bias1[8], w32v[8];
  #pragma unroll
  for (int j = 0; j < 2; ++j)
    #pragma unroll
    for (int g = 0; g < 4; ++g) {
      int n = (g * 32 + wave * 2 + j) * 16 + jl;
      bias0[j * 4 + g] = b0s[n];
      bias1[j * 4 + g] = b1s[n];
      w32v[j * 4 + g] = w32s[n];
    }

  // Input build: one element per thread, row = wave, col = lane (coalesced).
  const float* xbase = x + (size_t)(bbase + wave) * S_LEN * F_IN + lane;
  const int abuild = wave * STA + lane;

  __syncthreads();

  for (int t = 0; t < S_LEN; ++t) {
    const int tw = t / WEEK;
    const bool use_orig = (((tw & 1) == 0) && ((tw + 1) * WEEK <= S_LEN)) || (t == 0);
    const float flag = use_orig ? 0.f : 1.f;

    // ---- build cur = [feat0, x[:,1:32]] ----
    {
      float v = xbase[(size_t)t * F_IN];
      if (lane == 0 && !use_orig) v = prev_pred[wave];
      A[abuild] = f2bf(v);
    }
    __syncthreads();

    // ============ layer 0 GEMM: gates0 = [cur | h0_prev] @ W0^T + b0_eff ====
    v8f acc[8];
    #pragma unroll
    for (int i2 = 0; i2 < 8; ++i2) {
      float bv = bias0[i2] + flag * w32v[i2];   // fold flag column into bias
      v8f a;
      #pragma unroll
      for (int r = 0; r < 8; ++r) a[r] = bv;
      acc[i2] = a;
    }
    {
      const uint16_t* ap = &A[jl * STA + mhalf * 16];
      #pragma unroll 1
      for (int ks = 0; ks < NK0; ++ks) {
        v16bf af = *(const v16bf*)(ap + ks * 32);
        #pragma unroll
        for (int j = 0; j < 2; ++j) {
          #pragma unroll
          for (int g = 0; g < 4; ++g) {
            const int ntile = g * 32 + (wave * 2 + j);
            const v16bf bf = *(const v16bf*)(w0p + (((size_t)(ntile * NK0 + ks)) << 9) + (lane << 4));
            acc[j * 4 + g] = __builtin_amdgcn_wmma_f32_16x16x32_bf16(
                false, af, false, bf, (short)0, acc[j * 4 + g], false, false);
          }
        }
      }
    }
    __syncthreads();   // all waves done reading cur + h0_prev

    // ---- layer0 cell update (in-register), write h0 -> A cols 32..543 ----
    #pragma unroll
    for (int j = 0; j < 2; ++j) {
      const int col = COL_H0 + (wave * 2 + j) * 16 + jl;
      #pragma unroll
      for (int r = 0; r < 8; ++r) {
        float iv = acc[j * 4 + 0][r], fv = acc[j * 4 + 1][r];
        float gv = acc[j * 4 + 2][r], ov = acc[j * 4 + 3][r];
        float c = sigm(fv) * c0s[j][r] + sigm(iv) * tanh_fast(gv);
        c0s[j][r] = c;
        uint16_t hb = f2bf(sigm(ov) * tanh_fast(c));
        const int m = r + mhalf * 8;
        A[m * STA + col] = hb;
      }
    }
    __syncthreads();

    // ============ layer 1 GEMM: gates1 = [h0 | h1] @ W1^T + b1 ==============
    #pragma unroll
    for (int i2 = 0; i2 < 8; ++i2) {
      float bv = bias1[i2];
      v8f a;
      #pragma unroll
      for (int r = 0; r < 8; ++r) a[r] = bv;
      acc[i2] = a;
    }
    {
      const uint16_t* ap = &A[jl * STA + COL_H0 + mhalf * 16];
      #pragma unroll 1
      for (int ks = 0; ks < NK1; ++ks) {
        v16bf af = *(const v16bf*)(ap + ks * 32);
        #pragma unroll
        for (int j = 0; j < 2; ++j) {
          #pragma unroll
          for (int g = 0; g < 4; ++g) {
            const int ntile = g * 32 + (wave * 2 + j);
            const v16bf bf = *(const v16bf*)(w1p + (((size_t)(ntile * NK1 + ks)) << 9) + (lane << 4));
            acc[j * 4 + g] = __builtin_amdgcn_wmma_f32_16x16x32_bf16(
                false, af, false, bf, (short)0, acc[j * 4 + g], false, false);
          }
        }
      }
    }
    __syncthreads();   // all waves done reading A1 h1 region

    // ---- layer1 cell update, write h1 -> A cols 544..1055 ----
    #pragma unroll
    for (int j = 0; j < 2; ++j) {
      const int col = COL_H1 + (wave * 2 + j) * 16 + jl;
      #pragma unroll
      for (int r = 0; r < 8; ++r) {
        float iv = acc[j * 4 + 0][r], fv = acc[j * 4 + 1][r];
        float gv = acc[j * 4 + 2][r], ov = acc[j * 4 + 3][r];
        float c = sigm(fv) * c1s[j][r] + sigm(iv) * tanh_fast(gv);
        c1s[j][r] = c;
        uint16_t hb = f2bf(sigm(ov) * tanh_fast(c));
        const int m = r + mhalf * 8;
        A[m * STA + col] = hb;
      }
    }
    __syncthreads();

    // ---- output projection: out_t = h1 . Wout + bout, feed prev_pred ----
    {
      const int m = wave;   // one batch row per wave
      float s = 0.f;
      #pragma unroll
      for (int q = 0; q < 16; ++q) {
        const int cidx = lane + (q << 5);
        s += bf2f(A[m * STA + COL_H1 + cidx]) * Wout[cidx];
      }
      #pragma unroll
      for (int off2 = 16; off2 > 0; off2 >>= 1) s += __shfl_xor(s, off2, 32);
      if (lane == 0) {
        float o = s + bout[0];
        out[(size_t)(bbase + m) * S_LEN + t] = o;
        prev_pred[m] = o;
      }
    }
    __syncthreads();   // prev_pred visible before next step's build
  }
}

extern "C" void kernel_launch(void* const* d_in, const int* in_sizes, int n_in,
                              void* d_out, int out_size, void* d_ws, size_t ws_size,
                              hipStream_t stream)
{
  const float* x    = (const float*)d_in[0];
  const float* Wih0 = (const float*)d_in[1];
  const float* Whh0 = (const float*)d_in[2];
  const float* bih0 = (const float*)d_in[3];
  const float* bhh0 = (const float*)d_in[4];
  const float* Wih1 = (const float*)d_in[5];
  const float* Whh1 = (const float*)d_in[6];
  const float* bih1 = (const float*)d_in[7];
  const float* bhh1 = (const float*)d_in[8];
  const float* Wout = (const float*)d_in[9];
  const float* bout = (const float*)d_in[10];

  uint8_t* ws = (uint8_t*)d_ws;
  const size_t n0 = (size_t)NT * NK0 * 512;   // bf16 elems, layer0 packed
  const size_t n1 = (size_t)NT * NK1 * 512;   // bf16 elems, layer1 packed
  uint16_t* w0p  = (uint16_t*)ws;
  uint16_t* w1p  = (uint16_t*)(ws + n0 * 2);
  float*    b0s  = (float*)(ws + n0 * 2 + n1 * 2);
  float*    b1s  = b0s + NG;
  float*    w32s = b1s + NG;

  const int total = (int)(n0 + n1) + 3 * NG;
  pack_kernel<<<(total + 255) / 256, 256, 0, stream>>>(
      Wih0, Whh0, bih0, bhh0, Wih1, Whh1, bih1, bhh1, w0p, w1p, b0s, b1s, w32s);

  lstm_kernel<<<32, BLOCK, 0, stream>>>(
      x, w0p, w1p, b0s, b1s, w32s, Wout, bout, (float*)d_out);
}
